// RelationExtractionModel_63917703299456
// MI455X (gfx1250) — compile-verified, matrix operand order checked by reading, not run on hardware
//
#include <hip/hip_runtime.h>

// ---------------------------------------------------------------------------
// RelationExtraction model for MI455X (gfx1250, wave32, WMMA).
// Bulk GEMMs: v_wmma_f32_16x16x32_f16 (f16 operands, f32 accumulate).
// Sequential recurrences: persistent 32-wave workgroups, h state in LDS (f16),
// c / tree accumulators in f32. Unrolling capped in recurrent kernels to keep
// per-wave VGPR use ~60 and avoid scratch spills in the per-step hot loop.
// ---------------------------------------------------------------------------

#define BB 64
#define TT 512
#define BT (BB * TT)

typedef _Float16 f16;
typedef __attribute__((ext_vector_type(16))) _Float16 v16h;
typedef __attribute__((ext_vector_type(8)))  _Float16 v8h;
typedef __attribute__((ext_vector_type(8)))  float    v8f;

__device__ __forceinline__ float sigmoidf_(float x) { return 1.0f / (1.0f + __expf(-x)); }

// D = A(16x32) x B(32x16) + C, f32 accumulate
__device__ __forceinline__ v8f wmma32(v16h a, v16h b, v8f c) {
  return __builtin_amdgcn_wmma_f32_16x16x32_f16(
      /*neg_a=*/false, a, /*neg_b=*/false, b,
      /*c_mod=*/(short)0, c, /*reuse_a=*/false, /*reuse_b=*/false);
}

// Load one lane's share of a 16x32 f16 tile (works for A row-major [M,K] and
// for weights stored N-major [N,K]; identical per-lane pattern).
// ISA layout: lanes 0-15 hold rows 0-15; low half-wave K=0..7 & 16..23,
// high half-wave K=8..15 & 24..31 (f16 pairs per VGPR) -> two 16B loads.
__device__ __forceinline__ v16h load_tile16x32(const f16* __restrict__ base,
                                               int ld, int r0, int k0, int lane) {
  const int r    = r0 + (lane & 15);
  const int hsel = (lane >> 4) & 1;
  const f16* p = base + (size_t)r * ld + k0 + hsel * 8;
  v8h lo = *(const v8h*)(p);
  v8h hi = *(const v8h*)(p + 16);
  v16h out;
#pragma unroll
  for (int i = 0; i < 8; ++i) { out[i] = lo[i]; out[i + 8] = hi[i]; }
  return out;
}

// ---------------------------------------------------------------------------
// Weight repack: f32 [K,N] row-major -> f16 [N,Kpad] (zero pad K), so B tiles
// load with the same contiguous pattern as A tiles.
// ---------------------------------------------------------------------------
__global__ void pack_wt_kernel(const float* __restrict__ src, f16* __restrict__ dst,
                               int K, int N, int Kpad) {
  int idx = blockIdx.x * blockDim.x + threadIdx.x;
  if (idx >= N * Kpad) return;
  int n = idx / Kpad, k = idx - n * Kpad;
  dst[idx] = (k < K) ? (f16)src[(size_t)k * N + n] : (f16)0.0f;
}

// ---------------------------------------------------------------------------
// Build feature matrix (B*T, 928) f16: [phobert 768 | pos 32 | dep 32 |
// dist1 32 | dist2 32 | sdp 16 | zero-pad 16]
// ---------------------------------------------------------------------------
__global__ void build_feat_kernel(const float* __restrict__ phobert,
                                  const float* __restrict__ pos_tab,
                                  const float* __restrict__ dep_tab,
                                  const float* __restrict__ dist_tab,
                                  const float* __restrict__ sdp_tab,
                                  const int* __restrict__ pos_ids,
                                  const int* __restrict__ dep_ids,
                                  const int* __restrict__ sdp_pos,
                                  const int* __restrict__ e1_idx,
                                  const int* __restrict__ e2_idx,
                                  f16* __restrict__ featH) {
  const int F = 928;
  int idx = blockIdx.x * blockDim.x + threadIdx.x;
  if (idx >= BT * F) return;
  int tok = idx / F, f = idx - tok * F;
  int b = tok / TT, t = tok - b * TT;
  float v;
  if (f < 768)      v = phobert[(size_t)tok * 768 + f];
  else if (f < 800) v = pos_tab[pos_ids[tok] * 32 + (f - 768)];
  else if (f < 832) v = dep_tab[dep_ids[tok] * 32 + (f - 800)];
  else if (f < 864) { int d = t - e1_idx[b] + 10; d = d < 0 ? 0 : (d > 20 ? 20 : d);
                      v = dist_tab[d * 32 + (f - 832)]; }
  else if (f < 896) { int d = t - e2_idx[b] + 10; d = d < 0 ? 0 : (d > 20 ? 20 : d);
                      v = dist_tab[d * 32 + (f - 864)]; }
  else if (f < 912) v = sdp_tab[sdp_pos[tok] * 16 + (f - 896)];
  else              v = 0.0f;
  featH[idx] = (f16)v;
}

// dep embedding into tree_x columns [512,544)
__global__ void build_depe_kernel(const float* __restrict__ dep_tab,
                                  const int* __restrict__ dep_ids,
                                  f16* __restrict__ treeX) {
  int idx = blockIdx.x * blockDim.x + threadIdx.x;
  if (idx >= BT * 32) return;
  int tok = idx >> 5, j = idx & 31;
  treeX[(size_t)tok * 544 + 512 + j] = (f16)dep_tab[dep_ids[tok] * 32 + j];
}

// ---------------------------------------------------------------------------
// Bulk WMMA GEMM: C[M,N] = act(A[M,Kpad] @ Wt[N,Kpad]^T + bias)
// Each wave computes one 16x64 strip (A-tile register reuse across 4 N-tiles).
// ACT: 0 = linear, 1 = tanh. Cf (f32) and/or Ch (f16) outputs, shared ldc.
// ---------------------------------------------------------------------------
template <int ACT>
__global__ __launch_bounds__(256)
void wmma_gemm_kernel(const f16* __restrict__ A, int lda,
                      const f16* __restrict__ Wt, int ldw,
                      const float* __restrict__ bias,
                      float* __restrict__ Cf, f16* __restrict__ Ch, int ldc,
                      int M, int N, int K) {
  const int lane = threadIdx.x & 31;
  const int wave = threadIdx.x >> 5;
  const int stripsN = N >> 6;
  const int totalStrips = (M >> 4) * stripsN;
  const int strip = blockIdx.x * 8 + wave;
  if (strip >= totalStrips) return;  // wave-uniform
  const int tm = strip / stripsN;
  const int sn = strip - tm * stripsN;
  const int r0 = tm << 4, n0 = sn << 6;
  const int nc = lane & 15, hsel = (lane >> 4) & 1;

  v8f acc[4];
#pragma unroll
  for (int j = 0; j < 4; ++j) {
    float bv = bias ? bias[n0 + j * 16 + nc] : 0.0f;
#pragma unroll
    for (int i = 0; i < 8; ++i) acc[j][i] = bv;
  }
#pragma clang loop unroll_count(2)
  for (int k0 = 0; k0 < K; k0 += 32) {
    if (k0 + 32 < K)  // pull next K-chunk toward the caches (global_prefetch_b8)
      __builtin_prefetch(A + (size_t)(r0 + (lane & 15)) * lda + k0 + 32, 0, 1);
    v16h a = load_tile16x32(A, lda, r0, k0, lane);
#pragma unroll
    for (int j = 0; j < 4; ++j) {
      v16h b = load_tile16x32(Wt, ldw, n0 + j * 16, k0, lane);
      acc[j] = wmma32(a, b, acc[j]);
    }
  }
#pragma unroll
  for (int j = 0; j < 4; ++j)
#pragma unroll
    for (int i = 0; i < 8; ++i) {
      float v = acc[j][i];
      if (ACT == 1) v = tanhf(v);
      size_t o = (size_t)(r0 + i + hsel * 8) * ldc + n0 + j * 16 + nc;
      if (Cf) Cf[o] = v;
      if (Ch) Ch[o] = (f16)v;
    }
}

// ---------------------------------------------------------------------------
// BiLSTM: grid = 2 blocks (one per direction), 1024 threads = 32 waves/WGP.
// h (64x256) lives in LDS as f16; c in global f32. Input projections are
// precomputed (preG). Per step: G = preG[:,t] + h @ Whh^T via WMMA, then
// elementwise gate update. Writes h into tree_x columns [dir*256, dir*256+256).
// Unrolling disabled on strip/K loops: live set must stay < ~128 VGPRs at
// 32 waves/WGP, otherwise the compiler spills B-tiles to scratch in the
// 512-step critical path.
// ---------------------------------------------------------------------------
__global__ __launch_bounds__(1024)
void bilstm_kernel(const float* __restrict__ preGf, const float* __restrict__ preGb,
                   const f16* __restrict__ WhhFt, const f16* __restrict__ WhhBt,
                   float* __restrict__ cbuf,  // [2][64*256]
                   float* __restrict__ Gbuf,  // [2][64*1024]
                   f16* __restrict__ treeX) { // [B*T, 544]
  __shared__ __align__(16) f16 hS[BB * 256];
  const int dir = blockIdx.x;
  const float* preG = dir ? preGb : preGf;
  const f16* Wt = dir ? WhhBt : WhhFt;
  float* c = cbuf + (size_t)dir * BB * 256;
  float* G = Gbuf + (size_t)dir * BB * 1024;
  const int tid = threadIdx.x, lane = tid & 31, wave = tid >> 5;
  const int nc = lane & 15, hsel = (lane >> 4) & 1;

  for (int e = tid; e < BB * 256; e += 1024) { hS[e] = (f16)0.0f; c[e] = 0.0f; }
  __syncthreads();

  for (int s = 0; s < TT; ++s) {
    const int t = dir ? (TT - 1 - s) : s;
    // ---- G = preG[:,t,:] + hS @ Wt^T   (64x1024, K=256): 64 strips / 32 waves
#pragma clang loop unroll(disable)
    for (int strip = wave; strip < 64; strip += 32) {
      int r0 = (strip >> 4) << 4, n0 = (strip & 15) << 6;
      v8f acc[4];
#pragma unroll
      for (int j = 0; j < 4; ++j)
#pragma unroll
        for (int i = 0; i < 8; ++i) {
          int m = r0 + i + hsel * 8;
          acc[j][i] = preG[((size_t)m * TT + t) * 1024 + n0 + j * 16 + nc];
        }
#pragma clang loop unroll(disable)
      for (int k0 = 0; k0 < 256; k0 += 32) {
        v16h a = load_tile16x32(hS, 256, r0, k0, lane);
#pragma unroll
        for (int j = 0; j < 4; ++j) {
          v16h b = load_tile16x32(Wt, 256, n0 + j * 16, k0, lane);
          acc[j] = wmma32(a, b, acc[j]);
        }
      }
#pragma unroll
      for (int j = 0; j < 4; ++j)
#pragma unroll
        for (int i = 0; i < 8; ++i)
          G[(r0 + i + hsel * 8) * 1024 + n0 + j * 16 + nc] = acc[j][i];
    }
    __syncthreads();
    // ---- gate update
    for (int e = tid; e < BB * 256; e += 1024) {
      int b = e >> 8, j = e & 255;
      float gi = G[b * 1024 + j];
      float gf = G[b * 1024 + 256 + j];
      float gg = G[b * 1024 + 512 + j];
      float go = G[b * 1024 + 768 + j];
      float cn = sigmoidf_(gf) * c[e] + sigmoidf_(gi) * tanhf(gg);
      c[e] = cn;
      float h = sigmoidf_(go) * tanhf(cn);
      hS[e] = (f16)h;
      treeX[((size_t)b * TT + t) * 544 + dir * 256 + j] = (f16)h;
    }
    __syncthreads();
  }
}

// ---------------------------------------------------------------------------
// TreeLSTM: single block, 1024 threads. t = T-1 .. 0.
//   iou = iou_pre[:,t] + h_acc[:,t] @ Uiou^T   (WMMA 64x768, K=256)
//   c = sig(i)*tanh(u) + fc_acc[:,t]; h = sig(o)*tanh(c)
//   f = sig(fx_pre[b, parent] + h @ Uf^T)      (WMMA 64x256, K=256)
//   h_acc[b,parent] += h; fc_acc[b,parent] += f*c  (unique (b,n) per thread)
// ---------------------------------------------------------------------------
__global__ __launch_bounds__(1024)
void tree_kernel(const float* __restrict__ iou_pre, const float* __restrict__ fx_pre,
                 const f16* __restrict__ UiouT, const f16* __restrict__ UfT,
                 const int* __restrict__ dep_heads,
                 float* __restrict__ h_acc, float* __restrict__ fc_acc,
                 float* __restrict__ hs,
                 float* __restrict__ G,      // 64*768
                 float* __restrict__ cbuf) { // 64*256
  __shared__ __align__(16) f16 hS[BB * 256];
  const int tid = threadIdx.x, lane = tid & 31, wave = tid >> 5;
  const int nc = lane & 15, hsel = (lane >> 4) & 1;

  for (int s = 0; s < TT; ++s) {
    const int t = TT - 1 - s;
    // stage h_acc[:, t, :] as f16 for WMMA
    for (int e = tid; e < BB * 256; e += 1024) {
      int b = e >> 8, j = e & 255;
      hS[e] = (f16)h_acc[((size_t)b * TT + t) * 256 + j];
    }
    __syncthreads();
    // ---- iou GEMM (48 strips)
#pragma clang loop unroll(disable)
    for (int strip = wave; strip < 48; strip += 32) {
      int r0 = (strip / 12) << 4, n0 = (strip % 12) << 6;
      v8f acc[4];
#pragma unroll
      for (int j = 0; j < 4; ++j)
#pragma unroll
        for (int i = 0; i < 8; ++i) {
          int m = r0 + i + hsel * 8;
          acc[j][i] = iou_pre[((size_t)m * TT + t) * 768 + n0 + j * 16 + nc];
        }
#pragma clang loop unroll(disable)
      for (int k0 = 0; k0 < 256; k0 += 32) {
        v16h a = load_tile16x32(hS, 256, r0, k0, lane);
#pragma unroll
        for (int j = 0; j < 4; ++j) {
          v16h b = load_tile16x32(UiouT, 256, n0 + j * 16, k0, lane);
          acc[j] = wmma32(a, b, acc[j]);
        }
      }
#pragma unroll
      for (int j = 0; j < 4; ++j)
#pragma unroll
        for (int i = 0; i < 8; ++i)
          G[(r0 + i + hsel * 8) * 768 + n0 + j * 16 + nc] = acc[j][i];
    }
    __syncthreads();
    // ---- cell/hidden update
    for (int e = tid; e < BB * 256; e += 1024) {
      int b = e >> 8, j = e & 255;
      float gi = G[b * 768 + j];
      float go = G[b * 768 + 256 + j];
      float gu = G[b * 768 + 512 + j];
      float cc = sigmoidf_(gi) * tanhf(gu) + fc_acc[((size_t)b * TT + t) * 256 + j];
      float h  = sigmoidf_(go) * tanhf(cc);
      cbuf[e] = cc;
      hS[e] = (f16)h;
      hs[((size_t)b * TT + t) * 256 + j] = h;
    }
    __syncthreads();
    // ---- forget-gate GEMM (16 strips) + scatter to parents
#pragma clang loop unroll(disable)
    for (int strip = wave; strip < 16; strip += 32) {
      int r0 = (strip >> 2) << 4, n0 = (strip & 3) << 6;
      int pa[8];
#pragma unroll
      for (int i = 0; i < 8; ++i) pa[i] = dep_heads[(r0 + i + hsel * 8) * TT + t];
      v8f acc[4];
#pragma unroll
      for (int j = 0; j < 4; ++j)
#pragma unroll
        for (int i = 0; i < 8; ++i) {
          int m = r0 + i + hsel * 8;
          acc[j][i] = fx_pre[((size_t)m * TT + pa[i]) * 256 + n0 + j * 16 + nc];
        }
#pragma clang loop unroll(disable)
      for (int k0 = 0; k0 < 256; k0 += 32) {
        v16h a = load_tile16x32(hS, 256, r0, k0, lane);
#pragma unroll
        for (int j = 0; j < 4; ++j) {
          v16h b = load_tile16x32(UfT, 256, n0 + j * 16, k0, lane);
          acc[j] = wmma32(a, b, acc[j]);
        }
      }
#pragma unroll
      for (int j = 0; j < 4; ++j)
#pragma unroll
        for (int i = 0; i < 8; ++i) {
          int m = r0 + i + hsel * 8;
          int n = n0 + j * 16 + nc;
          float f = sigmoidf_(acc[j][i]);
          size_t dst = ((size_t)m * TT + pa[i]) * 256 + n;
          h_acc[dst]  += hs[((size_t)m * TT + t) * 256 + n];
          fc_acc[dst] += f * cbuf[m * 256 + n];
        }
    }
    __syncthreads();
  }
}

// ---------------------------------------------------------------------------
// Head: logits = relu(root_h @ W1 + b1) @ W2 + b2  (tiny; scalar math)
// ---------------------------------------------------------------------------
__global__ __launch_bounds__(1024)
void final_kernel(const float* __restrict__ hs, const int* __restrict__ root_idx,
                  const float* __restrict__ W1, const float* __restrict__ b1,
                  const float* __restrict__ W2, const float* __restrict__ b2,
                  float* __restrict__ hid, float* __restrict__ out) {
  const int tid = threadIdx.x;
  for (int e = tid; e < BB * 256; e += 1024) {
    int b = e >> 8, n = e & 255;
    const float* x = hs + ((size_t)b * TT + root_idx[b]) * 256;
    float acc = b1[n];
    for (int k = 0; k < 256; ++k) acc += x[k] * W1[k * 256 + n];
    hid[e] = fmaxf(acc, 0.0f);
  }
  __syncthreads();
  for (int e = tid; e < BB * 10; e += 1024) {
    int b = e / 10, n = e - b * 10;
    float acc = b2[n];
    for (int k = 0; k < 256; ++k) acc += hid[b * 256 + k] * W2[k * 10 + n];
    out[e] = acc;
  }
}

// ---------------------------------------------------------------------------
extern "C" void kernel_launch(void* const* d_in, const int* in_sizes, int n_in,
                              void* d_out, int out_size, void* d_ws, size_t ws_size,
                              hipStream_t stream) {
  (void)in_sizes; (void)n_in; (void)out_size; (void)ws_size;
  const float* phobert  = (const float*)d_in[0];
  const float* pos_tab  = (const float*)d_in[1];
  const float* dep_tab  = (const float*)d_in[2];
  const float* dist_tab = (const float*)d_in[3];
  const float* sdp_tab  = (const float*)d_in[4];
  const float* ft_W  = (const float*)d_in[5];
  const float* ft_b  = (const float*)d_in[6];
  const float* Wih_f = (const float*)d_in[7];
  const float* Whh_f = (const float*)d_in[8];
  const float* b_f   = (const float*)d_in[9];
  const float* Wih_b = (const float*)d_in[10];
  const float* Whh_b = (const float*)d_in[11];
  const float* b_b   = (const float*)d_in[12];
  const float* Wiou  = (const float*)d_in[13];
  const float* Uiou  = (const float*)d_in[14];
  const float* biou  = (const float*)d_in[15];
  const float* Wf    = (const float*)d_in[16];
  const float* Uf    = (const float*)d_in[17];
  const float* bf    = (const float*)d_in[18];
  const float* W1    = (const float*)d_in[19];
  const float* b1    = (const float*)d_in[20];
  const float* W2    = (const float*)d_in[21];
  const float* b2    = (const float*)d_in[22];
  const int* pos_ids   = (const int*)d_in[23];
  const int* dep_ids   = (const int*)d_in[24];
  const int* sdp_pos   = (const int*)d_in[25];
  const int* e1_idx    = (const int*)d_in[26];
  const int* e2_idx    = (const int*)d_in[27];
  const int* root_idx  = (const int*)d_in[28];
  const int* dep_heads = (const int*)d_in[29];
  float* out = (float*)d_out;

  // ---- workspace bump allocator (256B aligned), ~410 MB total with aliasing
  char* base = (char*)d_ws;
  size_t off = 0;
  auto alloc = [&](size_t bytes) -> char* {
    off = (off + 255) & ~(size_t)255;
    char* p = base + off;
    off += bytes;
    return p;
  };
  f16* ftWt  = (f16*)alloc((size_t)512  * 928 * 2);
  f16* WihFt = (f16*)alloc((size_t)1024 * 512 * 2);
  f16* WhhFt = (f16*)alloc((size_t)1024 * 256 * 2);
  f16* WihBt = (f16*)alloc((size_t)1024 * 512 * 2);
  f16* WhhBt = (f16*)alloc((size_t)1024 * 256 * 2);
  f16* WiouT = (f16*)alloc((size_t)768  * 544 * 2);
  f16* UiouT = (f16*)alloc((size_t)768  * 256 * 2);
  f16* WfT   = (f16*)alloc((size_t)256  * 544 * 2);
  f16* UfT   = (f16*)alloc((size_t)256  * 256 * 2);
  f16*   featH = (f16*)  alloc((size_t)BT * 928 * 2);
  f16*   xH    = (f16*)  alloc((size_t)BT * 512 * 2);
  float* preGf = (float*)alloc((size_t)BT * 1024 * 4);
  float* preGb = (float*)alloc((size_t)BT * 1024 * 4);
  f16*   treeX = (f16*)  alloc((size_t)BT * 544 * 2);
  float* Gb_lstm = (float*)alloc((size_t)2 * BB * 1024 * 4);
  float* cb_lstm = (float*)alloc((size_t)2 * BB * 256 * 4);
  float* G_tree  = (float*)alloc((size_t)BB * 768 * 4);
  float* c_tree  = (float*)alloc((size_t)BB * 256 * 4);
  float* hid     = (float*)alloc((size_t)BB * 256 * 4);
  // aliases (time-disjoint with preGf/preGb contents):
  float* iou_pre = preGf;                         // BT*768 f32 (<= BT*1024)
  float* fx_pre  = preGb;                         // BT*256
  float* h_acc   = preGb + (size_t)BT * 256;      // BT*256
  float* fc_acc  = preGb + (size_t)BT * 512;      // BT*256
  float* hs      = preGb + (size_t)BT * 768;      // BT*256

  auto pack = [&](const float* src, f16* dst, int K, int N, int Kpad) {
    int total = N * Kpad;
    pack_wt_kernel<<<(total + 255) / 256, 256, 0, stream>>>(src, dst, K, N, Kpad);
  };
  pack(ft_W,  ftWt,  912, 512,  928);
  pack(Wih_f, WihFt, 512, 1024, 512);
  pack(Whh_f, WhhFt, 256, 1024, 256);
  pack(Wih_b, WihBt, 512, 1024, 512);
  pack(Whh_b, WhhBt, 256, 1024, 256);
  pack(Wiou,  WiouT, 544, 768,  544);
  pack(Uiou,  UiouT, 256, 768,  256);
  pack(Wf,    WfT,   544, 256,  544);
  pack(Uf,    UfT,   256, 256,  256);

  build_feat_kernel<<<((size_t)BT * 928 + 255) / 256, 256, 0, stream>>>(
      phobert, pos_tab, dep_tab, dist_tab, sdp_tab,
      pos_ids, dep_ids, sdp_pos, e1_idx, e2_idx, featH);
  build_depe_kernel<<<((size_t)BT * 32 + 255) / 256, 256, 0, stream>>>(
      dep_tab, dep_ids, treeX);

  auto gemm = [&](int act, const f16* A, int lda, const f16* Wt, int ldw,
                  const float* bias, float* Cf, f16* Ch, int ldc,
                  int M, int N, int K) {
    int strips = (M / 16) * (N / 64);
    int grid = (strips + 7) / 8;
    if (act)
      wmma_gemm_kernel<1><<<grid, 256, 0, stream>>>(A, lda, Wt, ldw, bias, Cf, Ch, ldc, M, N, K);
    else
      wmma_gemm_kernel<0><<<grid, 256, 0, stream>>>(A, lda, Wt, ldw, bias, Cf, Ch, ldc, M, N, K);
  };

  // x = tanh(feat @ ft_W + ft_b) -> f16
  gemm(1, featH, 928, ftWt, 928, ft_b, nullptr, xH, 512, BT, 512, 928);
  // input projections for both LSTM directions -> f32 pre-gates
  gemm(0, xH, 512, WihFt, 512, b_f, preGf, nullptr, 1024, BT, 1024, 512);
  gemm(0, xH, 512, WihBt, 512, b_b, preGb, nullptr, 1024, BT, 1024, 512);

  bilstm_kernel<<<2, 1024, 0, stream>>>(preGf, preGb, WhhFt, WhhBt,
                                        cb_lstm, Gb_lstm, treeX);

  // TreeLSTM input-side precomputes (aliases are now safe: preG dead)
  gemm(0, treeX, 544, WiouT, 544, biou, iou_pre, nullptr, 768, BT, 768, 544);
  gemm(0, treeX, 544, WfT,   544, bf,   fx_pre,  nullptr, 256, BT, 256, 544);

  hipMemsetAsync(h_acc, 0, (size_t)BT * 512 * 4, stream);  // h_acc + fc_acc

  tree_kernel<<<1, 1024, 0, stream>>>(iou_pre, fx_pre, UiouT, UfT, dep_heads,
                                      h_acc, fc_acc, hs, G_tree, c_tree);

  final_kernel<<<1, 1024, 0, stream>>>(hs, root_idx, W1, b1, W2, b2, hid, out);
}
